// GATLayer_26259430048439
// MI455X (gfx1250) — compile-verified, matrix operand order checked by reading, not run on hardware
//
#include <hip/hip_runtime.h>
#include <hip/hip_bf16.h>
#include <math.h>

// ---------------------------------------------------------------------------
// GAT layer for MI455X (gfx1250, wave32, WMMA).
//
// Decomposition:
//   z   = X @ W1.T, z_i = X @ W2.T           (WMMA GEMM, 16-row tiles/wave)
//   s1  = z . Wa[0:64], s2 = z . Wa[64:128]  (per-node scalars -> edge logits
//                                             become s1[src]+s2[dst]+t*Wa[128])
//   segment softmax over dst via float atomicMax / atomicAdd
//   z_neighbor scatter: 64 lanes per edge, coalesced fp32 atomicAdd
//   h = relu(z_i + z_neighbor)
// ---------------------------------------------------------------------------

typedef __attribute__((ext_vector_type(2))) float v2f;
typedef __attribute__((ext_vector_type(8))) float v8f;

#define HAS_WMMA_F32 __has_builtin(__builtin_amdgcn_wmma_f32_16x16x4_f32)

// ---------------- Kernel 1: dual GEMM via WMMA ----------------
// One wave (32 threads) computes a 16x64 strip of both z and z_i.
__global__ __launch_bounds__(32) void gat_gemm_kernel(
    const float* __restrict__ X,    // [N,128]
    const float* __restrict__ W1,   // [64,128]
    const float* __restrict__ W2,   // [64,128]
    float* __restrict__ z,          // [N,64]
    float* __restrict__ zi,         // [N,64]
    int N)
{
    const int rowbase = blockIdx.x * 16;
    const int lane    = threadIdx.x;       // 0..31
    const int half    = lane >> 4;         // lane group (0 / 1)
    const int l16     = lane & 15;

    v8f acc1[4] = {v8f{}, v8f{}, v8f{}, v8f{}};
    v8f acc2[4] = {v8f{}, v8f{}, v8f{}, v8f{}};

    int arow = rowbase + l16;
    if (arow >= N) arow = N - 1;           // clamp (uniform-safe, no EXEC holes)
    const float* xrow = X + (size_t)arow * 128;

#if HAS_WMMA_F32
    // A: 16x4 f32 tile. lane<16 holds row=l16, K={k0,k0+1}; lane>=16 K={k0+2,k0+3}
    for (int k0 = 0; k0 < 128; k0 += 4) {
        const int kk = k0 + 2 * half;
        v2f a; a.x = xrow[kk]; a.y = xrow[kk + 1];
#pragma unroll
        for (int j = 0; j < 4; ++j) {
            const int col = j * 16 + l16;       // output feature
            v2f b1, b2;
            b1.x = W1[col * 128 + kk]; b1.y = W1[col * 128 + kk + 1];
            b2.x = W2[col * 128 + kk]; b2.y = W2[col * 128 + kk + 1];
            acc1[j] = __builtin_amdgcn_wmma_f32_16x16x4_f32(
                false, a, false, b1, (short)0, acc1[j], false, false);
            acc2[j] = __builtin_amdgcn_wmma_f32_16x16x4_f32(
                false, a, false, b2, (short)0, acc2[j], false, false);
        }
    }
#else
    // Fallback: probe-confirmed bf16 WMMA (fp32 accumulate).
    typedef __attribute__((ext_vector_type(16))) __bf16 v16bf;
    for (int k0 = 0; k0 < 128; k0 += 32) {
        v16bf a;
#pragma unroll
        for (int e = 0; e < 16; ++e) {
            const int v = e >> 1, hb = e & 1;
            const int K = ((v < 4) ? 2 * v : 16 + 2 * (v - 4)) + hb + 8 * half;
            a[e] = (__bf16)xrow[k0 + K];
        }
#pragma unroll
        for (int j = 0; j < 4; ++j) {
            const int col = j * 16 + l16;
            v16bf b1, b2;
#pragma unroll
            for (int e = 0; e < 16; ++e) {
                const int v = e >> 1, hb = e & 1;
                const int K = ((v < 4) ? 2 * v : 16 + 2 * (v - 4)) + hb + 8 * half;
                b1[e] = (__bf16)W1[col * 128 + k0 + K];
                b2[e] = (__bf16)W2[col * 128 + k0 + K];
            }
            acc1[j] = __builtin_amdgcn_wmma_f32_16x16x32_bf16(
                false, a, false, b1, (short)0, acc1[j], false, false);
            acc2[j] = __builtin_amdgcn_wmma_f32_16x16x32_bf16(
                false, a, false, b2, (short)0, acc2[j], false, false);
        }
    }
#endif

    // C/D layout: VGPR r, lanes 0-15 -> row r, lanes 16-31 -> row r+8
#pragma unroll
    for (int j = 0; j < 4; ++j) {
#pragma unroll
        for (int r = 0; r < 8; ++r) {
            const int row = rowbase + r + 8 * half;
            if (row < N) {
                z [(size_t)row * 64 + j * 16 + l16] = acc1[j][r];
                zi[(size_t)row * 64 + j * 16 + l16] = acc2[j][r];
            }
        }
    }
}

// ---------------- Kernel 2: per-node attention scalars + init ----------------
// One wave per node: s1 = z.Wa[0:64], s2 = z.Wa[64:128]; init m=-inf, denom=0,
// zero the z_neighbor accumulator (== d_out).
__global__ __launch_bounds__(256) void gat_node_kernel(
    const float* __restrict__ z, const float* __restrict__ Wa,
    float* __restrict__ s1, float* __restrict__ s2,
    float* __restrict__ m, float* __restrict__ denom,
    float* __restrict__ acc, int N)
{
    const int wave = (blockIdx.x * blockDim.x + threadIdx.x) >> 5;
    const int lane = threadIdx.x & 31;
    if (wave >= N) return;
    const float* zr = z + (size_t)wave * 64;
    float a0 = zr[lane], a1 = zr[lane + 32];
    float p1 = a0 * Wa[lane]      + a1 * Wa[lane + 32];
    float p2 = a0 * Wa[64 + lane] + a1 * Wa[96 + lane];
#pragma unroll
    for (int off = 16; off > 0; off >>= 1) {
        p1 += __shfl_xor(p1, off, 32);
        p2 += __shfl_xor(p2, off, 32);
    }
    acc[(size_t)wave * 64 + lane]      = 0.0f;
    acc[(size_t)wave * 64 + lane + 32] = 0.0f;
    if (lane == 0) {
        s1[wave]    = p1;
        s2[wave]    = p2;
        m[wave]     = -__builtin_inff();
        denom[wave] = 0.0f;
    }
}

// ---------------- Kernel 3: edge logits + segment max ----------------
__global__ __launch_bounds__(256) void gat_edge_logit_kernel(
    const float* __restrict__ edge_d,
    const int* __restrict__ src, const int* __restrict__ dst,
    const float* __restrict__ s1, const float* __restrict__ s2,
    const float* __restrict__ W0, const float* __restrict__ Wa,
    float* __restrict__ ebuf, float* __restrict__ m, int E)
{
    const int e = blockIdx.x * blockDim.x + threadIdx.x;
    if (e >= E) return;
    const int d = dst[e];
    float v = s1[src[e]] + s2[d] + edge_d[e] * W0[0] * Wa[128];
    v = (v > 0.0f) ? v : 0.01f * v;          // leaky_relu, torch default slope
    ebuf[e] = v;
    // float atomic-max via int/uint punning (correct across mixed signs)
    if (v >= 0.0f) atomicMax((int*)(m + d), __float_as_int(v));
    else           atomicMin((unsigned int*)(m + d), __float_as_uint(v));
}

// ---------------- Kernel 4: exp + segment sum ----------------
__global__ __launch_bounds__(256) void gat_edge_exp_kernel(
    const int* __restrict__ dst, const float* __restrict__ m,
    float* __restrict__ ebuf, float* __restrict__ denom, int E)
{
    const int e = blockIdx.x * blockDim.x + threadIdx.x;
    if (e >= E) return;
    const int d = dst[e];
    const float ex = __expf(ebuf[e] - m[d]);
    ebuf[e] = ex;                             // reuse buffer in place
    atomicAdd(denom + d, ex);
}

// ---------------- Kernel 5: weighted scatter of source messages ----------------
// 64 lanes per edge -> coalesced 256B gather of z[src] and 256B atomicAdd burst.
__global__ __launch_bounds__(256) void gat_scatter_kernel(
    const int* __restrict__ src, const int* __restrict__ dst,
    const float* __restrict__ ebuf, const float* __restrict__ denom,
    const float* __restrict__ z, float* __restrict__ acc, int E)
{
    const int e = blockIdx.x * 4 + (threadIdx.x >> 6);
    const int c = threadIdx.x & 63;
    if (e >= E) return;
    const int d = dst[e];
    const float alpha = ebuf[e] / denom[d];
    atomicAdd(acc + (size_t)d * 64 + c, alpha * z[(size_t)src[e] * 64 + c]);
}

// ---------------- Kernel 6: h = relu(z_i + z_neighbor) ----------------
__global__ __launch_bounds__(256) void gat_final_kernel(
    const float* __restrict__ zi, float* __restrict__ out, int total)
{
    const int i = blockIdx.x * blockDim.x + threadIdx.x;
    if (i >= total) return;
    const float v = zi[i] + out[i];
    out[i] = (v > 0.0f) ? v : 0.0f;
}

// ---------------------------------------------------------------------------
extern "C" void kernel_launch(void* const* d_in, const int* in_sizes, int n_in,
                              void* d_out, int out_size, void* d_ws, size_t ws_size,
                              hipStream_t stream)
{
    const float* X   = (const float*)d_in[0];   // node_feats [N,128]
    const float* ed  = (const float*)d_in[1];   // edge_d [E,1]
    const float* W0  = (const float*)d_in[2];   // [1,1]
    const float* W1  = (const float*)d_in[3];   // [64,128]
    const float* W2  = (const float*)d_in[4];   // [64,128]
    const float* Wa  = (const float*)d_in[5];   // [1,129]
    const int*   src = (const int*)d_in[6];     // [E]
    const int*   dst = (const int*)d_in[7];     // [E]

    const int N = in_sizes[0] / 128;
    const int E = in_sizes[6];

    float* ws    = (float*)d_ws;
    float* z     = ws;  ws += (size_t)N * 64;
    float* zi    = ws;  ws += (size_t)N * 64;
    float* s1    = ws;  ws += N;
    float* s2    = ws;  ws += N;
    float* m     = ws;  ws += N;
    float* denom = ws;  ws += N;
    float* ebuf  = ws;  ws += E;
    float* out   = (float*)d_out;               // doubles as z_neighbor accumulator

    const int rowTiles = (N + 15) / 16;
    gat_gemm_kernel<<<rowTiles, 32, 0, stream>>>(X, W1, W2, z, zi, N);

    const int nodeThreads = N * 32;
    gat_node_kernel<<<(nodeThreads + 255) / 256, 256, 0, stream>>>(
        z, Wa, s1, s2, m, denom, out, N);

    gat_edge_logit_kernel<<<(E + 255) / 256, 256, 0, stream>>>(
        ed, src, dst, s1, s2, W0, Wa, ebuf, m, E);

    gat_edge_exp_kernel<<<(E + 255) / 256, 256, 0, stream>>>(
        dst, m, ebuf, denom, E);

    gat_scatter_kernel<<<(E + 3) / 4, 256, 0, stream>>>(
        src, dst, ebuf, denom, z, out, E);

    gat_final_kernel<<<(N * 64 + 255) / 256, 256, 0, stream>>>(zi, out, N * 64);
}
